// SecretRQVAE_54073638256949
// MI455X (gfx1250) — matrix operand order repaired and checked by reading, not compiled, not run on hardware
//
#include <hip/hip_runtime.h>
#include <hip/hip_bf16.h>

typedef __attribute__((ext_vector_type(16))) _Float16 v16h;
typedef __attribute__((ext_vector_type(8)))  _Float16 v8h;
typedef __attribute__((ext_vector_type(8)))  float    v8f;

// ---------------------------------------------------------------------------
// CDNA5 async global->LDS copy (ASYNCcnt path), per CDNA5 ISA
// ---------------------------------------------------------------------------
__device__ __forceinline__ void async_load_b128(void* lds, const void* g) {
    unsigned l = (unsigned)(size_t)lds;              // LDS byte address (low 32b)
    unsigned long long a = (unsigned long long)(size_t)g;
    asm volatile("global_load_async_to_lds_b128 %0, %1, off"
                 :: "v"(l), "v"(a) : "memory");
}
__device__ __forceinline__ void wait_async0() {
    asm volatile("s_wait_asynccnt 0x0" ::: "memory");
}

// ---------------------------------------------------------------------------
// One-shot weight packing: f32 (OIHW or torch-transposed IOHW) -> f16 GEMM rows
//   wpk[m][kg], m padded to MP (mult of 64, zero rows), kg padded to KtotP
//   (mult of 32, zero cols).  16B-aligned rows -> async b128 staging later.
// ---------------------------------------------------------------------------
__global__ __launch_bounds__(256)
void pack_weights(const float* __restrict__ w, _Float16* __restrict__ wpk,
                  int Cout, int Cin, int KK, int KtotP, int MP, int transposed)
{
    int idx = blockIdx.x * 256 + threadIdx.x;
    if (idx >= MP * KtotP) return;
    int m = idx / KtotP, kg = idx - m * KtotP;
    int Ktot = Cin * KK;
    float v = 0.f;
    if (m < Cout && kg < Ktot) {
        if (transposed) { int ci = kg / KK, rc = kg - ci * KK; v = w[(ci * Cout + m) * KK + rc]; }
        else            v = w[m * Ktot + kg];
    }
    wpk[idx] = (_Float16)v;
}

// ---------------------------------------------------------------------------
// Implicit-GEMM convolution / transposed convolution.
//   GEMM: M = Cout (padded), K = Cin*KSZ*KSZ (padded), N = B*Hout*Wout
//   Block = 256 threads (8 waves), tile = 64(M) x 64(N).
//   A tile: async global->LDS b128 copies of prepacked f16 weights.
//   B tile: im2col gather with per-thread hoisted n-decode.
// ---------------------------------------------------------------------------
template<int KSZ, int STRIDE, int PAD, bool TRANSPOSED, bool RELU, bool TANH>
__global__ __launch_bounds__(256)
void conv_gemm(const float* __restrict__ in, const _Float16* __restrict__ wpk,
               const float* __restrict__ bias, float* __restrict__ out,
               int B, int Cin, int Cout, int Hin, int Win, int Hout, int Wout,
               int KtotP)
{
    constexpr int KK = KSZ * KSZ;
    const int Ktot = Cin * KK;
    const int Ntot = B * Hout * Wout;

    __shared__ __align__(16) _Float16 As[64][32];   // [m][k]
    __shared__ __align__(16) _Float16 Bs[64][32];   // [n][k] (transposed im2col)

    const int tid  = threadIdx.x;
    const int lane = tid & 31;
    const int wave = tid >> 5;
    const int half = lane >> 4;
    const int lmod = lane & 15;
    const int wN = wave & 3;    // N subtile 0..3
    const int wM = wave >> 2;   // M pair    0..1

    const int nBase = blockIdx.x * 64;
    const int mBase = blockIdx.y * 64;

    // ---- hoisted per-thread B-gather coordinates: fixed kk lane, 8 fixed rows
    const int kkF = tid & 31;
    const int nr0 = tid >> 5;
    int oyA[8], oxA[8]; size_t baseA[8]; bool nvA[8];
    #pragma unroll
    for (int j = 0; j < 8; ++j) {
        int n = nBase + nr0 + j * 8;
        nvA[j] = (n < Ntot);
        int nn = nvA[j] ? n : 0;
        int ox = nn % Wout; int t2 = nn / Wout;
        oxA[j] = ox; oyA[j] = t2 % Hout;
        baseA[j] = (size_t)(t2 / Hout) * Cin * Hin * Win;
    }
    // async A staging: each thread owns one 16B segment of the 4KB tile
    const int aRow = tid >> 2, aSeg = tid & 3;

    v8f acc0 = {}; v8f acc1 = {};

    const int nKchunks = KtotP >> 5;
    for (int kc = 0; kc < nKchunks; ++kc) {
        const int kk0 = kc << 5;

        // ---- A tile: contiguous async copy from packed f16 weights ----
        async_load_b128(&As[aRow][aSeg * 8],
                        wpk + (size_t)(mBase + aRow) * KtotP + kk0 + aSeg * 8);

        // ---- B tile: im2col gather (one kg decode per thread per chunk) ----
        {
            int kg = kk0 + kkF;
            float v[8];
            if (kg < Ktot) {
                int ci = kg / KK; int rc = kg - ci * KK;
                int r  = rc / KSZ; int c = rc - r * KSZ;
                size_t cioff = (size_t)ci * Hin * Win;
                #pragma unroll
                for (int j = 0; j < 8; ++j) {
                    float vv = 0.f;
                    if (nvA[j]) {
                        if (TRANSPOSED) {
                            int ty = oyA[j] + PAD - r, tx = oxA[j] + PAD - c;
                            if (ty >= 0 && tx >= 0 && (ty % STRIDE) == 0 && (tx % STRIDE) == 0) {
                                int iy = ty / STRIDE, ix = tx / STRIDE;
                                if (iy < Hin && ix < Win)
                                    vv = in[baseA[j] + cioff + (size_t)iy * Win + ix];
                            }
                        } else {
                            int iy = oyA[j] * STRIDE - PAD + r, ix = oxA[j] * STRIDE - PAD + c;
                            if (iy >= 0 && iy < Hin && ix >= 0 && ix < Win)
                                vv = in[baseA[j] + cioff + (size_t)iy * Win + ix];
                        }
                    }
                    v[j] = vv;
                }
            } else {
                #pragma unroll
                for (int j = 0; j < 8; ++j) v[j] = 0.f;
            }
            #pragma unroll
            for (int j = 0; j < 8; ++j) Bs[nr0 + j * 8][kkF] = (_Float16)v[j];
        }
        wait_async0();
        __syncthreads();

        // B fragment: n = wN*16+lmod, K run = half*16 .. +16 (contiguous 32B)
        v16h bfr;
        {
            const v8h* p = (const v8h*)&Bs[wN * 16 + lmod][half * 16];
            ((v8h*)&bfr)[0] = p[0];
            ((v8h*)&bfr)[1] = p[1];
        }
        // A fragments: rows m, K run half*16 .. +16 (contiguous)
        v16h a0, a1;
        {
            int r0 = (wM * 2 + 0) * 16 + lmod;
            ((v8h*)&a0)[0] = *(const v8h*)&As[r0][half * 16];
            ((v8h*)&a0)[1] = *(const v8h*)&As[r0][half * 16 + 8];
            int r1 = (wM * 2 + 1) * 16 + lmod;
            ((v8h*)&a1)[0] = *(const v8h*)&As[r1][half * 16];
            ((v8h*)&a1)[1] = *(const v8h*)&As[r1][half * 16 + 8];
        }
        acc0 = __builtin_amdgcn_wmma_f32_16x16x32_f16(false, a0, false, bfr, (short)0, acc0, false, false);
        acc1 = __builtin_amdgcn_wmma_f32_16x16x32_f16(false, a1, false, bfr, (short)0, acc1, false, false);
        __syncthreads();
    }

    // ---- epilogue: C/D layout lane->N, vgpr r -> M = half*8 + r ----
    const int nCol = nBase + wN * 16 + lmod;
    if (nCol < Ntot) {
        int ox = nCol % Wout; int t = nCol / Wout;
        int oy = t % Hout;    int b = t / Hout;
        #pragma unroll
        for (int s = 0; s < 2; ++s) {
            v8f acc = s ? acc1 : acc0;
            int mSub = (wM * 2 + s) * 16;
            #pragma unroll
            for (int r = 0; r < 8; ++r) {
                int m = mBase + mSub + half * 8 + r;
                if (m < Cout) {
                    float v = acc[r] + bias[m];
                    if (RELU) v = v > 0.f ? v : 0.f;
                    if (TANH) v = tanhf(v);
                    out[(((size_t)b * Cout + m) * Hout + oy) * Wout + ox] = v;
                }
            }
        }
    }
}

// ---------------------------------------------------------------------------
// Codebook prep: f32 -> f16 copy + per-code squared norm; zero loss slots.
// ---------------------------------------------------------------------------
__global__ __launch_bounds__(256)
void prep_codes(const float* __restrict__ cb, _Float16* __restrict__ cbh,
                float* __restrict__ norms, float* __restrict__ loss)
{
    const int code = blockIdx.x;
    const int d    = threadIdx.x;
    float v = cb[(size_t)code * 256 + d];
    cbh[(size_t)code * 256 + d] = (_Float16)v;
    float s = v * v;
    __shared__ float red[8];
    for (int off = 16; off > 0; off >>= 1) s += __shfl_down(s, off, 32);
    if ((threadIdx.x & 31) == 0) red[threadIdx.x >> 5] = s;
    __syncthreads();
    if (threadIdx.x < 8) {
        float t = red[threadIdx.x];
        for (int off = 4; off > 0; off >>= 1) t += __shfl_down(t, off, 32);
        if (threadIdx.x == 0) norms[code] = t;
    }
    if (blockIdx.x == 0 && threadIdx.x < 4) loss[threadIdx.x] = 0.f;
}

// ---------------------------------------------------------------------------
// Residual VQ: one wave owns 16 tokens; distances via WMMA (codes x residual^T),
// in-wave argmin, straight-through update, commit-loss atomics.
// ---------------------------------------------------------------------------
__global__ __launch_bounds__(32)
void rvq_kernel(const float* __restrict__ z, const float* __restrict__ cb,
                const _Float16* __restrict__ cbh, const float* __restrict__ norms,
                float* __restrict__ quant, float* __restrict__ idxOut,
                float* __restrict__ loss)
{
    __shared__ __align__(16) float    resf[16][256];
    __shared__ __align__(16) _Float16 resh[16][256];
    __shared__ __align__(16) float    qout[16][256];
    __shared__ int bestIdx[16];

    const int lane = threadIdx.x;
    const int half = lane >> 4, lmod = lane & 15;
    const int tokBase = blockIdx.x * 16;

    {
        int t = lane >> 1, d0 = (lane & 1) * 128;
        int tok = tokBase + t;
        int b = tok >> 12, y = (tok >> 6) & 63, x = tok & 63;
        const float* zp = z + (size_t)b * 256 * 4096 + (size_t)y * 64 + x;
        for (int d = 0; d < 128; ++d) {
            float v = zp[(size_t)(d0 + d) * 4096];
            resf[t][d0 + d] = v;
            resh[t][d0 + d] = (_Float16)v;
            qout[t][d0 + d] = 0.f;
        }
    }
    __syncthreads();

    const float lossScale = 1.0f / (65536.0f * 256.0f);

    for (int q = 0; q < 4; ++q) {
        const _Float16* cq = cbh + (size_t)q * 1024 * 256;
        float bestV = 3.4e38f; int bestI = 0;

        v16h bfr[8];
        #pragma unroll
        for (int kc = 0; kc < 8; ++kc) {
            const v8h* p = (const v8h*)&resh[lmod][kc * 32 + half * 16];
            ((v8h*)&bfr[kc])[0] = p[0];
            ((v8h*)&bfr[kc])[1] = p[1];
        }

        for (int mt = 0; mt < 64; ++mt) {
            v8f accd = {};
            const _Float16* arow = cq + (size_t)(mt * 16 + lmod) * 256;
            #pragma unroll
            for (int kc = 0; kc < 8; ++kc) {
                v16h afr;
                ((v8h*)&afr)[0] = *(const v8h*)(arow + kc * 32 + half * 16);
                ((v8h*)&afr)[1] = *(const v8h*)(arow + kc * 32 + half * 16 + 8);
                accd = __builtin_amdgcn_wmma_f32_16x16x32_f16(false, afr, false, bfr[kc], (short)0, accd, false, false);
            }
            #pragma unroll
            for (int r = 0; r < 8; ++r) {
                int code = mt * 16 + half * 8 + r;
                float d = norms[q * 1024 + code] - 2.0f * accd[r];
                if (d < bestV) { bestV = d; bestI = code; }
            }
        }
        {
            float ov = __shfl_xor(bestV, 16, 32);
            int   oi = __shfl_xor(bestI, 16, 32);
            if (ov < bestV || (ov == bestV && oi < bestI)) { bestV = ov; bestI = oi; }
        }
        if (half == 0) bestIdx[lmod] = bestI;
        __syncthreads();

        float ls = 0.f;
        {
            int t = lane >> 1, d0 = (lane & 1) * 128;
            const float* crow = cb + ((size_t)q * 1024 + bestIdx[t]) * 256;
            for (int d = 0; d < 128; ++d) {
                float c  = crow[d0 + d];
                float r  = resf[t][d0 + d];
                float df = c - r;
                ls += df * df;
                float nr = r - c;
                resf[t][d0 + d] = nr;
                resh[t][d0 + d] = (_Float16)nr;
                qout[t][d0 + d] += c;
            }
        }
        for (int off = 16; off > 0; off >>= 1) ls += __shfl_down(ls, off, 32);
        if (lane == 0) atomicAdd(&loss[q], ls * lossScale);

        if (half == 0) {
            int tok = tokBase + lmod;
            int b = tok >> 12, y = (tok >> 6) & 63, x = tok & 63;
            idxOut[(((size_t)b * 4 + q) * 64 + y) * 64 + x] = (float)bestI;
        }
        __syncthreads();
    }

    {
        int t = lane >> 1, d0 = (lane & 1) * 128;
        int tok = tokBase + t;
        int b = tok >> 12, y = (tok >> 6) & 63, x = tok & 63;
        float* qp = quant + (size_t)b * 256 * 4096 + (size_t)y * 64 + x;
        for (int d = 0; d < 128; ++d)
            qp[(size_t)(d0 + d) * 4096] = qout[t][d0 + d];
    }
}

// ---------------------------------------------------------------------------
extern "C" void kernel_launch(void* const* d_in, const int* in_sizes, int n_in,
                              void* d_out, int out_size, void* d_ws, size_t ws_size,
                              hipStream_t stream) {
    const float* x      = (const float*)d_in[0];
    const float* enc_w1 = (const float*)d_in[1];  const float* enc_b1 = (const float*)d_in[2];
    const float* enc_w2 = (const float*)d_in[3];  const float* enc_b2 = (const float*)d_in[4];
    const float* enc_w3 = (const float*)d_in[5];  const float* enc_b3 = (const float*)d_in[6];
    const float* enc_w4 = (const float*)d_in[7];  const float* enc_b4 = (const float*)d_in[8];
    const float* cbooks = (const float*)d_in[9];
    const float* dec_w1 = (const float*)d_in[10]; const float* dec_b1 = (const float*)d_in[11];
    const float* dec_w2 = (const float*)d_in[12]; const float* dec_b2 = (const float*)d_in[13];
    const float* dec_w3 = (const float*)d_in[14]; const float* dec_b3 = (const float*)d_in[15];
    const float* dec_w4 = (const float*)d_in[16]; const float* dec_b4 = (const float*)d_in[17];

    // workspace layout (bytes)
    char* ws = (char*)d_ws;
    float*    Z     = (float*)(ws + 0);                      //  67,108,864 B  (16,256,64,64)
    float*    Abuf  = (float*)(ws + 67108864);               //  67,108,864 B  (h1 / d2)
    float*    Bbuf  = (float*)(ws + 134217728);              //  33,554,432 B  (h2 / d1)
    float*    Cbuf  = (float*)(ws + 167772160);              //  67,108,864 B  (h3)
    float*    D3    = (float*)(ws + 234881024);              // 268,435,456 B  (d3)
    _Float16* cbh   = (_Float16*)(ws + 503316480);           //   2,097,152 B
    float*    norms = (float*)(ws + 505413632);              //      16,384 B
    _Float16* wpB   = (_Float16*)(ws + 505430016);           //   3,153,920 B packed weights

    // packed weight regions (halfword offsets)
    _Float16* wp1 = wpB + 0;        // 64   x 64
    _Float16* wp2 = wpB + 4096;     // 128  x 1024
    _Float16* wp3 = wpB + 135168;   // 256  x 1152
    _Float16* wp4 = wpB + 430080;   // 256  x 2304
    _Float16* wd1 = wpB + 1019904;  // 128  x 2304
    _Float16* wd2 = wpB + 1314816;  // 64   x 2048  (transposed layout)
    _Float16* wd3 = wpB + 1445888;  // 64   x 1024  (transposed layout)
    _Float16* wd4 = wpB + 1511424;  // 64   x 1024  (transposed layout, M padded 3->64)

    // d_out layout (floats): recon | indices | commit_loss | quantized
    float* recon  = (float*)d_out;
    float* idxOut = recon + 12582912;
    float* lossO  = idxOut + 262144;
    float* quant  = lossO + 4;

    // one-shot prep: codebooks + packed weights
    prep_codes<<<4096, 256, 0, stream>>>(cbooks, cbh, norms, lossO);
    pack_weights<<<(64   * 64   + 255) / 256, 256, 0, stream>>>(enc_w1, wp1,  64,   3, 16,   64,  64, 0);
    pack_weights<<<(128  * 1024 + 255) / 256, 256, 0, stream>>>(enc_w2, wp2, 128,  64, 16, 1024, 128, 0);
    pack_weights<<<(256  * 1152 + 255) / 256, 256, 0, stream>>>(enc_w3, wp3, 256, 128,  9, 1152, 256, 0);
    pack_weights<<<(256  * 2304 + 255) / 256, 256, 0, stream>>>(enc_w4, wp4, 256, 256,  9, 2304, 256, 0);
    pack_weights<<<(128  * 2304 + 255) / 256, 256, 0, stream>>>(dec_w1, wd1, 128, 256,  9, 2304, 128, 0);
    pack_weights<<<(64   * 2048 + 255) / 256, 256, 0, stream>>>(dec_w2, wd2,  64, 128, 16, 2048,  64, 1);
    pack_weights<<<(64   * 1024 + 255) / 256, 256, 0, stream>>>(dec_w3, wd3,  64,  64, 16, 1024,  64, 1);
    pack_weights<<<(64   * 1024 + 255) / 256, 256, 0, stream>>>(dec_w4, wd4,   3,  64, 16, 1024,  64, 1);

    // encoder
    conv_gemm<4,2,1,false,true ,false><<<dim3(4096, 1), 256, 0, stream>>>(x,     wp1, enc_b1, Abuf, 16,   3,  64, 256, 256, 128, 128,   64);
    conv_gemm<4,2,1,false,true ,false><<<dim3(1024, 2), 256, 0, stream>>>(Abuf,  wp2, enc_b2, Bbuf, 16,  64, 128, 128, 128,  64,  64, 1024);
    conv_gemm<3,1,1,false,true ,false><<<dim3(1024, 4), 256, 0, stream>>>(Bbuf,  wp3, enc_b3, Cbuf, 16, 128, 256,  64,  64,  64,  64, 1152);
    conv_gemm<3,1,1,false,false,false><<<dim3(1024, 4), 256, 0, stream>>>(Cbuf,  wp4, enc_b4, Z,    16, 256, 256,  64,  64,  64,  64, 2304);

    // residual VQ: 65536 tokens / 16 per wave
    rvq_kernel<<<4096, 32, 0, stream>>>(Z, cbooks, cbh, norms, quant, idxOut, lossO);

    // decoder
    conv_gemm<3,1,1,false,true ,false><<<dim3(1024,  2), 256, 0, stream>>>(quant, wd1, dec_b1, Bbuf,  16, 256, 128,  64,  64,  64,  64, 2304);
    conv_gemm<4,2,1,true ,true ,false><<<dim3(4096,  1), 256, 0, stream>>>(Bbuf,  wd2, dec_b2, Abuf,  16, 128,  64,  64,  64, 128, 128, 2048);
    conv_gemm<4,2,1,true ,true ,false><<<dim3(16384, 1), 256, 0, stream>>>(Abuf,  wd3, dec_b3, D3,    16,  64,  64, 128, 128, 256, 256, 1024);
    conv_gemm<4,2,1,true ,false,true ><<<dim3(65536, 1), 256, 0, stream>>>(D3,    wd4, dec_b4, recon, 16,  64,   3, 256, 256, 512, 512, 1024);
}
